// GPTModel_16509854286386
// MI455X (gfx1250) — compile-verified, matrix operand order checked by reading, not run on hardware
//
#include <hip/hip_runtime.h>
#include <hip/hip_bf16.h>
#include <math.h>

// ---------------- model dims (match reference) ----------------
#define Vv   32000
#define Dd   1024
#define Hh   16
#define Ll   6
#define FFf  4096
#define Bb   2
#define Ss   1024
#define HDd  (Dd / Hh)      // 64
#define NT   (Bb * Ss)      // 2048 tokens

typedef long long i64;
typedef __bf16 bf16;

// ---------------- WMMA types ----------------
typedef __attribute__((ext_vector_type(16))) __bf16 v16bf;
typedef __attribute__((ext_vector_type(8)))  __bf16 v8bf;   // 16 bytes
typedef __attribute__((ext_vector_type(8)))  float  v8f;

union frag_u { v16bf v; v8bf h[2]; };

static __device__ __forceinline__ bf16 f2bf(float f) {
    union { float f; unsigned u; } x; x.f = f;
    unsigned r = x.u + 0x7FFFu + ((x.u >> 16) & 1u);   // round-to-nearest-even
    unsigned short hv = (unsigned short)(r >> 16);
    union { unsigned short s; bf16 b; } y; y.s = hv;
    return y.b;
}

// ---------------- CDNA5 async global->LDS copy (ASYNCcnt path) ----------------
#if defined(__AMDGCN__) && __has_builtin(__builtin_amdgcn_global_load_async_to_lds_b128)
#define CDNA5_ASYNC 1
#else
#define CDNA5_ASYNC 0
#endif

// exact builtin parameter type per hipcc diagnostic:
//   int __vector(4) in AS1 (global) / AS3 (LDS)
typedef int b128v __attribute__((vector_size(16)));
#if defined(__AMDGCN__)
typedef __attribute__((address_space(1))) b128v* gp128_t;
typedef __attribute__((address_space(3))) b128v* lp128_t;
#endif

static __device__ __forceinline__ void cp16_g2l(const void* gsrc, void* ldst) {
#if CDNA5_ASYNC
    __builtin_amdgcn_global_load_async_to_lds_b128(
        (gp128_t)gsrc, (lp128_t)ldst, 0, 0);
#else
    *(uint4*)ldst = *(const uint4*)gsrc;
#endif
}
static __device__ __forceinline__ void async_wait_all() {
#if CDNA5_ASYNC
#if __has_builtin(__builtin_amdgcn_s_wait_asynccnt)
    __builtin_amdgcn_s_wait_asynccnt(0);
#else
    asm volatile("s_wait_asynccnt 0" ::: "memory");
#endif
#endif
}

// =====================================================================
// Batched bf16 WMMA GEMM, LDS double-buffered:
//   C[z] = alpha * (A[z] @ Bt[z]^T) + bias
//   A  : bf16 [M,K] row-major, lda.  Bt : bf16 [N,K] row-major, ldb.
//   C  : f32 out (C!=null)  OR  Cbf : bf16 out.
//   Block tile 128(M) x 128(N), K step 32, double-buffered in LDS
//   (A/B tiles 128x32 bf16, padded row stride 40 elems -> 40KB total).
//   8 waves tiled 4(M) x 2(N); wave computes 32x64 = 8 WMMAs / K-step
//   from 12 ds_load_b128. Staging via async global->LDS (ASYNCcnt).
//   Requires M%128==0, K%32==0; N edge handled by zero-fill + store guard.
// =====================================================================
#define LDK 40      // padded LDS row stride (elements)
#define TSZ 5120    // 128*LDK elements per tile

__global__ __launch_bounds__(256)
void gemm_bf16_wmma(const bf16* __restrict__ A, const bf16* __restrict__ Bt,
                    const float* __restrict__ bias,
                    float* __restrict__ C, bf16* __restrict__ Cbf,
                    int M, int N, int K, int lda, int ldb, int ldc,
                    float alpha, int nH,
                    i64 sAb, i64 sAh, i64 sBb, i64 sBh, i64 sCb, i64 sCh)
{
    __shared__ bf16 smem[2][2][TSZ];   // [buf][A|B][tile]

    const int z  = blockIdx.z;
    const int zb = z / nH, zh = z % nH;
    const i64 coff = (i64)zb * sCb + (i64)zh * sCh;

    const int m00 = blockIdx.y * 128;
    const int n00 = blockIdx.x * 128;
    int rowsB = N - n00; if (rowsB > 128) rowsB = 128;

    const bf16* Ag = A  + (i64)zb * sAb + (i64)zh * sAh + (i64)m00 * lda;
    const bf16* Bg = Bt + (i64)zb * sBb + (i64)zh * sBh + (i64)n00 * ldb;

    const int tid  = threadIdx.x;
    const int lane = tid & 31;
    const int wave = tid >> 5;
    const int wm = wave & 3;            // 0..3 -> rows wm*32..+31
    const int wn = wave >> 2;           // 0..1 -> cols wn*64..+63

    const int srow = tid >> 2;          // staging: 0..63
    const int sseg = (tid & 3) * 8;     // staging: element offset 0/8/16/24

    const int r    = lane & 15;
    const int koff = (lane >> 4) * 8;

    v8f acc[4][2];
#pragma unroll
    for (int ni = 0; ni < 4; ++ni)
#pragma unroll
        for (int mi = 0; mi < 2; ++mi)
            acc[ni][mi] = (v8f){0.f,0.f,0.f,0.f,0.f,0.f,0.f,0.f};

    // ---- prologue: stage K-slice 0 into buffer 0 ----
    {
        bf16* As = &smem[0][0][0];
        bf16* Bs = &smem[0][1][0];
        cp16_g2l(Ag + (i64)srow * lda + sseg,        As + srow * LDK + sseg);
        cp16_g2l(Ag + (i64)(srow + 64) * lda + sseg, As + (srow + 64) * LDK + sseg);
        if (srow < rowsB)      cp16_g2l(Bg + (i64)srow * ldb + sseg,        Bs + srow * LDK + sseg);
        else                   *(uint4*)(Bs + srow * LDK + sseg) = make_uint4(0,0,0,0);
        if (srow + 64 < rowsB) cp16_g2l(Bg + (i64)(srow + 64) * ldb + sseg, Bs + (srow + 64) * LDK + sseg);
        else                   *(uint4*)(Bs + (srow + 64) * LDK + sseg) = make_uint4(0,0,0,0);
    }

    for (int k0 = 0; k0 < K; k0 += 32) {
        const int p = (k0 >> 5) & 1;
        async_wait_all();
        __syncthreads();

        // ---- stage next K-slice into the other buffer ----
        if (k0 + 32 < K) {
            const int kn = k0 + 32;
            bf16* As = &smem[p ^ 1][0][0];
            bf16* Bs = &smem[p ^ 1][1][0];
            cp16_g2l(Ag + (i64)srow * lda + kn + sseg,        As + srow * LDK + sseg);
            cp16_g2l(Ag + (i64)(srow + 64) * lda + kn + sseg, As + (srow + 64) * LDK + sseg);
            if (srow < rowsB)      cp16_g2l(Bg + (i64)srow * ldb + kn + sseg,        Bs + srow * LDK + sseg);
            else                   *(uint4*)(Bs + srow * LDK + sseg) = make_uint4(0,0,0,0);
            if (srow + 64 < rowsB) cp16_g2l(Bg + (i64)(srow + 64) * ldb + kn + sseg, Bs + (srow + 64) * LDK + sseg);
            else                   *(uint4*)(Bs + (srow + 64) * LDK + sseg) = make_uint4(0,0,0,0);
        }

        // ---- compute from buffer p ----
        const bf16* As = &smem[p][0][0];
        const bf16* Bs = &smem[p][1][0];

        frag_u a0, a1;
        const int ar = wm * 32 + r;
        a0.h[0] = *reinterpret_cast<const v8bf*>(As + ar * LDK + koff);
        a0.h[1] = *reinterpret_cast<const v8bf*>(As + ar * LDK + koff + 16);
        a1.h[0] = *reinterpret_cast<const v8bf*>(As + (ar + 16) * LDK + koff);
        a1.h[1] = *reinterpret_cast<const v8bf*>(As + (ar + 16) * LDK + koff + 16);

#pragma unroll
        for (int ni = 0; ni < 4; ++ni) {
            frag_u b;
            const int br = wn * 64 + ni * 16 + r;
            b.h[0] = *reinterpret_cast<const v8bf*>(Bs + br * LDK + koff);
            b.h[1] = *reinterpret_cast<const v8bf*>(Bs + br * LDK + koff + 16);
            acc[ni][0] = __builtin_amdgcn_wmma_f32_16x16x32_bf16(
                             false, a0.v, false, b.v, (short)0, acc[ni][0], false, false);
            acc[ni][1] = __builtin_amdgcn_wmma_f32_16x16x32_bf16(
                             false, a1.v, false, b.v, (short)0, acc[ni][1], false, false);
        }
    }

    // ---- epilogue: 32-bit C/D 16x16 layout; guard columns vs N ----
    const int cmBase = (lane >> 4) * 8;
    const int cL     = lane & 15;
#pragma unroll
    for (int ni = 0; ni < 4; ++ni) {
        const int cn = n00 + wn * 64 + ni * 16 + cL;
        if (cn >= N) continue;
        const float bvv = bias ? bias[cn] : 0.f;
#pragma unroll
        for (int mi = 0; mi < 2; ++mi) {
            const int mrow = m00 + wm * 32 + mi * 16 + cmBase;
#pragma unroll
            for (int v = 0; v < 8; ++v) {
                const float val = alpha * acc[ni][mi][v] + bvv;
                const i64 off = (i64)(mrow + v) * ldc + cn;
                if (Cbf) (Cbf + coff)[off] = f2bf(val);
                else     (C   + coff)[off] = val;
            }
        }
    }
}

// =====================================================================
// transpose + convert to bf16:  src[z] (T)[R,C] ld=lds  ->  dst[z] bf16 [C,R]
// =====================================================================
template <typename T>
__global__ __launch_bounds__(256)
void transpose_cvt_kernel(const T* __restrict__ src, bf16* __restrict__ dst,
                          int R, int C, int lds, int nH,
                          i64 sSb, i64 sSh, i64 sDz)
{
    __shared__ float tile[32][33];
    const int z = blockIdx.z;
    src += (i64)(z / nH) * sSb + (i64)(z % nH) * sSh;
    dst += (i64)z * sDz;
    const int c0 = blockIdx.x * 32, r0 = blockIdx.y * 32;
#pragma unroll
    for (int j = 0; j < 4; ++j) {
        const int rr = r0 + threadIdx.y + j * 8, cc = c0 + threadIdx.x;
        if (rr < R && cc < C)
            tile[threadIdx.y + j * 8][threadIdx.x] = (float)src[(i64)rr * lds + cc];
    }
    __syncthreads();
#pragma unroll
    for (int j = 0; j < 4; ++j) {
        const int cc = c0 + threadIdx.y + j * 8, rr = r0 + threadIdx.x;
        if (cc < C && rr < R)
            dst[(i64)cc * R + rr] = f2bf(tile[threadIdx.x][threadIdx.y + j * 8]);
    }
}

// =====================================================================
// block reductions (wave32 shuffles + LDS across 8 waves)
// =====================================================================
__device__ __forceinline__ float blockReduceMax(float v, float* sm) {
    for (int off = 16; off; off >>= 1) v = fmaxf(v, __shfl_xor(v, off, 32));
    const int wave = threadIdx.x >> 5, lane = threadIdx.x & 31;
    if (lane == 0) sm[wave] = v;
    __syncthreads();
    if (wave == 0) {
        float x = (lane < 8) ? sm[lane] : -INFINITY;
        for (int off = 4; off; off >>= 1) x = fmaxf(x, __shfl_xor(x, off, 32));
        if (lane == 0) sm[0] = x;
    }
    __syncthreads();
    float r = sm[0];
    __syncthreads();
    return r;
}
__device__ __forceinline__ float blockReduceSum(float v, float* sm) {
    for (int off = 16; off; off >>= 1) v += __shfl_xor(v, off, 32);
    const int wave = threadIdx.x >> 5, lane = threadIdx.x & 31;
    if (lane == 0) sm[wave] = v;
    __syncthreads();
    if (wave == 0) {
        float x = (lane < 8) ? sm[lane] : 0.f;
        for (int off = 4; off; off >>= 1) x += __shfl_xor(x, off, 32);
        if (lane == 0) sm[0] = x;
    }
    __syncthreads();
    float r = sm[0];
    __syncthreads();
    return r;
}

// =====================================================================
// embedding * sqrt(D) + sinusoidal positional encoding  (f32 residual)
// =====================================================================
__global__ __launch_bounds__(256)
void embed_kernel(const float* __restrict__ emb, const int* __restrict__ ids,
                  float* __restrict__ x)
{
    const i64 idx = (i64)blockIdx.x * blockDim.x + threadIdx.x;
    if (idx >= (i64)NT * Dd) return;
    const int t = (int)(idx / Dd);
    const int d = (int)(idx % Dd);
    const int pos = t % Ss;
    const int id  = ids[t];
    const int i2  = (d >> 1) << 1;
    const float freq = __expf(-9.2103403719761836f * (float)i2 / (float)Dd);
    const float ang  = (float)pos * freq;
    const float pe   = (d & 1) ? cosf(ang) : sinf(ang);
    x[idx] = emb[(i64)id * Dd + d] * 32.0f /* sqrt(1024) */ + pe;
}

// =====================================================================
// LayerNorm over D=1024, one block per token; writes bf16 (GEMM A operand)
// =====================================================================
__global__ __launch_bounds__(256)
void layernorm_kernel(const float* __restrict__ x, const float* __restrict__ g,
                      const float* __restrict__ b, bf16* __restrict__ y)
{
    __shared__ float sm[8];
    const float* xr = x + (i64)blockIdx.x * Dd;
    bf16*        yr = y + (i64)blockIdx.x * Dd;
    float s = 0.f, ss = 0.f;
    for (int i = threadIdx.x; i < Dd; i += 256) {
        const float v = xr[i]; s += v; ss += v * v;
    }
    s  = blockReduceSum(s, sm);
    ss = blockReduceSum(ss, sm);
    const float mean = s * (1.0f / Dd);
    const float var  = ss * (1.0f / Dd) - mean * mean;
    const float rstd = rsqrtf(var + 1e-5f);
    for (int i = threadIdx.x; i < Dd; i += 256)
        yr[i] = f2bf((xr[i] - mean) * rstd * g[i] + b[i]);
}

// =====================================================================
// causal softmax: reads f32 scores row, writes bf16 probabilities
// =====================================================================
__global__ __launch_bounds__(256)
void softmax_causal_kernel(const float* __restrict__ scores, bf16* __restrict__ probs)
{
    __shared__ float sm[8];
    const int q = blockIdx.x;
    const float* row = scores + ((i64)blockIdx.y * Ss + q) * Ss;
    bf16*        orow = probs + ((i64)blockIdx.y * Ss + q) * Ss;
    float m = -INFINITY;
    for (int k = threadIdx.x; k <= q; k += 256) m = fmaxf(m, row[k]);
    m = blockReduceMax(m, sm);
    float s = 0.f;
    float e[4];  // S/256 = 4 slots max
    int  nk = 0;
    for (int k = threadIdx.x; k <= q; k += 256) { e[nk] = __expf(row[k] - m); s += e[nk]; ++nk; }
    s = blockReduceSum(s, sm);
    const float inv = 1.0f / s;
    nk = 0;
    for (int k = threadIdx.x; k < Ss; k += 256) {
        const float p = (k <= q) ? e[nk] * inv : 0.f;
        if (k <= q) ++nk;
        orow[k] = f2bf(p);
    }
}

// =====================================================================
// elementwise: exact GELU f32 -> bf16 ; residual add (f32, in-place)
// =====================================================================
__global__ __launch_bounds__(256)
void gelu_cvt_kernel(const float* __restrict__ x, bf16* __restrict__ y, i64 n)
{
    const i64 i = (i64)blockIdx.x * blockDim.x + threadIdx.x;
    if (i < n) {
        const float v = x[i];
        y[i] = f2bf(0.5f * v * (1.0f + erff(v * 0.70710678118654752f)));
    }
}
__global__ __launch_bounds__(256)
void add_inplace_kernel(float* __restrict__ x, const float* __restrict__ y, i64 n)
{
    const i64 i = (i64)blockIdx.x * blockDim.x + threadIdx.x;
    if (i < n) x[i] += y[i];
}

// =====================================================================
// host-side helpers
// =====================================================================
static void launch_gemm(hipStream_t st,
                        const bf16* A, const bf16* Bt, const float* bias,
                        float* C, bf16* Cbf,
                        int M, int N, int K, int lda, int ldb, int ldc,
                        float alpha, int batB, int nH,
                        i64 sAb, i64 sAh, i64 sBb, i64 sBh, i64 sCb, i64 sCh)
{
    dim3 grid((N + 127) / 128, (M + 127) / 128, batB * nH);
    gemm_bf16_wmma<<<grid, dim3(256), 0, st>>>(A, Bt, bias, C, Cbf,
                                               M, N, K, lda, ldb, ldc,
                                               alpha, nH, sAb, sAh, sBb, sBh, sCb, sCh);
}
static void launch_transpose_f32(hipStream_t st, const float* src, bf16* dst,
                                 int R, int C, int lds, int nZ, i64 sSb, i64 sDz)
{
    dim3 grid((C + 31) / 32, (R + 31) / 32, nZ);
    transpose_cvt_kernel<float><<<grid, dim3(32, 8), 0, st>>>(src, dst, R, C, lds,
                                                              1, sSb, 0, sDz);
}

extern "C" void kernel_launch(void* const* d_in, const int* in_sizes, int n_in,
                              void* d_out, int out_size, void* d_ws, size_t ws_size,
                              hipStream_t stream)
{
    (void)in_sizes; (void)n_in; (void)out_size; (void)ws_size;
    const float* emb   = (const float*)d_in[0];
    const float* Wq    = (const float*)d_in[1];
    const float* bq    = (const float*)d_in[2];
    const float* Wk    = (const float*)d_in[3];
    const float* bk    = (const float*)d_in[4];
    const float* Wv    = (const float*)d_in[5];
    const float* bv    = (const float*)d_in[6];
    const float* Wo    = (const float*)d_in[7];
    const float* bo    = (const float*)d_in[8];
    const float* ln1g  = (const float*)d_in[9];
    const float* ln1b  = (const float*)d_in[10];
    const float* W1    = (const float*)d_in[11];
    const float* b1    = (const float*)d_in[12];
    const float* W2    = (const float*)d_in[13];
    const float* b2    = (const float*)d_in[14];
    const float* ln2g  = (const float*)d_in[15];
    const float* ln2b  = (const float*)d_in[16];
    const float* lnfg  = (const float*)d_in[17];
    const float* lnfb  = (const float*)d_in[18];
    const float* Wout  = (const float*)d_in[19];
    const int*   ids   = (const int*)d_in[20];
    float* out = (float*)d_out;

    // ---- workspace carve ----
    const i64 ND = (i64)NT * Dd;          // 2M
    const i64 NF = (i64)NT * FFf;         // 8M
    const i64 SC = (i64)Bb * Hh * Ss * Ss;// 32M

    float* x   = (float*)d_ws;            // residual stream (f32)
    float* t1  = x  + ND;                 // projection / FFN2 out (f32)
    float* sc  = t1 + ND;                 // attention scores (f32)
    float* ff  = sc + SC;                 // FFN1 pre-activation (f32)

    bf16* hB   = (bf16*)(ff + NF);        // LN output
    bf16* qB   = hB  + ND;
    bf16* kB   = qB  + ND;
    bf16* vB   = kB  + ND;
    bf16* vT   = vB  + ND;                // [B,H,HD,S]
    bf16* t0B  = vT  + ND;                // attention output [B,S,H,HD]
    bf16* scB  = t0B + ND;                // probabilities [B,H,S,S]
    bf16* ffB  = scB + SC;                // GELU output [NT,FF]

    bf16* WqT  = ffB + NF;                // [L][D,D] transposed
    bf16* WkT  = WqT + (i64)Ll * Dd * Dd;
    bf16* WvT  = WkT + (i64)Ll * Dd * Dd;
    bf16* WoT  = WvT + (i64)Ll * Dd * Dd;
    bf16* W1T  = WoT + (i64)Ll * Dd * Dd; // [L][FF,D]
    bf16* W2T  = W1T + (i64)Ll * Dd * FFf;// [L][D,FF]
    bf16* WoutT= W2T + (i64)Ll * FFf * Dd;// [V,D]

    const int eg = (int)((ND + 255) / 256);
    const float scale = 1.0f / sqrtf((float)HDd);

    // ---- one-time (per call) weight transpose-converts ----
    launch_transpose_f32(stream, Wq, WqT, Dd, Dd, Dd, Ll, (i64)Dd * Dd, (i64)Dd * Dd);
    launch_transpose_f32(stream, Wk, WkT, Dd, Dd, Dd, Ll, (i64)Dd * Dd, (i64)Dd * Dd);
    launch_transpose_f32(stream, Wv, WvT, Dd, Dd, Dd, Ll, (i64)Dd * Dd, (i64)Dd * Dd);
    launch_transpose_f32(stream, Wo, WoT, Dd, Dd, Dd, Ll, (i64)Dd * Dd, (i64)Dd * Dd);
    launch_transpose_f32(stream, W1, W1T, Dd, FFf, FFf, Ll, (i64)Dd * FFf, (i64)Dd * FFf);
    launch_transpose_f32(stream, W2, W2T, FFf, Dd, Dd, Ll, (i64)FFf * Dd, (i64)FFf * Dd);
    launch_transpose_f32(stream, Wout, WoutT, Dd, Vv, Vv, 1, 0, 0);

    // ---- embedding ----
    embed_kernel<<<eg, 256, 0, stream>>>(emb, ids, x);

    for (int l = 0; l < Ll; ++l) {
        const bf16* WqT_l = WqT + (i64)l * Dd * Dd;
        const bf16* WkT_l = WkT + (i64)l * Dd * Dd;
        const bf16* WvT_l = WvT + (i64)l * Dd * Dd;
        const bf16* WoT_l = WoT + (i64)l * Dd * Dd;
        const bf16* W1T_l = W1T + (i64)l * Dd * FFf;
        const bf16* W2T_l = W2T + (i64)l * FFf * Dd;

        // LN1 -> bf16
        layernorm_kernel<<<NT, 256, 0, stream>>>(x, ln1g + (i64)l * Dd, ln1b + (i64)l * Dd, hB);

        // Q,K,V projections (bf16 out)
        launch_gemm(stream, hB, WqT_l, bq + (i64)l * Dd, nullptr, qB,
                    NT, Dd, Dd, Dd, Dd, Dd, 1.f, 1, 1, 0,0, 0,0, 0,0);
        launch_gemm(stream, hB, WkT_l, bk + (i64)l * Dd, nullptr, kB,
                    NT, Dd, Dd, Dd, Dd, Dd, 1.f, 1, 1, 0,0, 0,0, 0,0);
        launch_gemm(stream, hB, WvT_l, bv + (i64)l * Dd, nullptr, vB,
                    NT, Dd, Dd, Dd, Dd, Dd, 1.f, 1, 1, 0,0, 0,0, 0,0);

        // V per-head transpose: [S,HD] -> [HD,S]  (bf16 -> bf16)
        {
            dim3 grid((HDd + 31) / 32, (Ss + 31) / 32, Bb * Hh);
            transpose_cvt_kernel<bf16><<<grid, dim3(32, 8), 0, stream>>>(
                vB, vT, Ss, HDd, Dd, Hh,
                (i64)Ss * Dd, (i64)HDd, (i64)HDd * Ss);
        }

        // scores[b,h] = scale * q_bh [S,HD] @ (k_bh [S,HD])^T  (f32 out)
        launch_gemm(stream, qB, kB, nullptr, sc, nullptr,
                    Ss, Ss, HDd, Dd, Dd, Ss, scale, Bb, Hh,
                    (i64)Ss * Dd, (i64)HDd,
                    (i64)Ss * Dd, (i64)HDd,
                    (i64)Hh * Ss * Ss, (i64)Ss * Ss);

        // causal softmax -> bf16 probabilities
        softmax_causal_kernel<<<dim3(Ss, Bb * Hh), 256, 0, stream>>>(sc, scB);

        // attn[b,h] = P [S,S] @ v_bh [S,HD]   (Bt = vT [HD,S]) -> bf16 [B,S,H,HD]
        launch_gemm(stream, scB, vT, nullptr, nullptr, t0B,
                    Ss, HDd, Ss, Ss, Ss, Dd, 1.f, Bb, Hh,
                    (i64)Hh * Ss * Ss, (i64)Ss * Ss,
                    (i64)Hh * HDd * Ss, (i64)HDd * Ss,
                    (i64)Ss * Dd, (i64)HDd);

        // output projection (f32) + residual
        launch_gemm(stream, t0B, WoT_l, bo + (i64)l * Dd, t1, nullptr,
                    NT, Dd, Dd, Dd, Dd, Dd, 1.f, 1, 1, 0,0, 0,0, 0,0);
        add_inplace_kernel<<<eg, 256, 0, stream>>>(x, t1, ND);

        // LN2 + FFN
        layernorm_kernel<<<NT, 256, 0, stream>>>(x, ln2g + (i64)l * Dd, ln2b + (i64)l * Dd, hB);
        launch_gemm(stream, hB, W1T_l, b1 + (i64)l * FFf, ff, nullptr,
                    NT, FFf, Dd, Dd, Dd, FFf, 1.f, 1, 1, 0,0, 0,0, 0,0);
        gelu_cvt_kernel<<<(int)((NF + 255) / 256), 256, 0, stream>>>(ff, ffB, NF);
        launch_gemm(stream, ffB, W2T_l, b2 + (i64)l * Dd, t1, nullptr,
                    NT, Dd, FFf, FFf, FFf, Dd, 1.f, 1, 1, 0,0, 0,0, 0,0);
        add_inplace_kernel<<<eg, 256, 0, stream>>>(x, t1, ND);
    }

    // final LN + logits
    layernorm_kernel<<<NT, 256, 0, stream>>>(x, lnfg, lnfb, hB);
    launch_gemm(stream, hB, WoutT, nullptr, out, nullptr,
                NT, Vv, Dd, Dd, Dd, Vv, 1.f, 1, 1, 0,0, 0,0, 0,0);
}
